// Head_25683904430270
// MI455X (gfx1250) — matrix operand order repaired and checked by reading, not compiled, not run on hardware
//
#include <hip/hip_runtime.h>

typedef __attribute__((ext_vector_type(16))) _Float16 v16h;
typedef __attribute__((ext_vector_type(8)))  _Float16 v8h;
typedef __attribute__((ext_vector_type(8)))  float    v8f;

#define Tn 256
#define Cn 384
#define Hn 64

// LDS element (f16) offsets
#define WT_OFF 0                 // [3][64][384]  transposed weights (Wq pre-scaled)
#define K_OFF  73728             // [256][64] row-major
#define Q_OFF  90112             // [256][64] row-major
#define V_OFF  106496            // [64][256] V transposed
#define P_OFF  122880            // [8 waves][16][32] P staging
#define LDS_BYTES (126976u * 2u) // 253952 B  (< 320 KB)

#if defined(__has_builtin)
#if __has_builtin(__builtin_amdgcn_sched_group_barrier)
#define SCHED_GROUP(mask, size, id) __builtin_amdgcn_sched_group_barrier(mask, size, id)
#endif
#endif
#ifndef SCHED_GROUP
#define SCHED_GROUP(mask, size, id)
#endif
// masks: 0x008 = MFMA/WMMA, 0x100 = DS read

static __device__ __forceinline__ v8f wmma16(const v16h a, const v16h b, v8f c) {
  return __builtin_amdgcn_wmma_f32_16x16x32_f16(false, a, false, b, (short)0, c,
                                                false, false);
}

static __device__ __forceinline__ v16h cat8(const v8h lo, const v8h hi8) {
  return __builtin_shufflevector(lo, hi8, 0, 1, 2, 3, 4, 5, 6, 7,
                                 8, 9, 10, 11, 12, 13, 14, 15);
}

__global__ __launch_bounds__(256, 1)
void Head_attn_kernel(const float* __restrict__ x, const float* __restrict__ Wk,
                      const float* __restrict__ Wq, const float* __restrict__ Wv,
                      float* __restrict__ out) {
  extern __shared__ _Float16 lds[];
  _Float16* wT = lds + WT_OFF;
  _Float16* kr = lds + K_OFF;
  _Float16* qr = lds + Q_OFF;
  _Float16* vt = lds + V_OFF;
  _Float16* ps = lds + P_OFF;

  const int tid  = threadIdx.x;
  const int lane = tid & 31;
  const int wv   = tid >> 5;
  const int n16  = lane & 15;   // column (B/C) or row (A) index within tile
  const int hi   = lane >> 4;   // half-wave select
  const int b    = blockIdx.x;
  const float* xb = x + (size_t)b * (Tn * Cn);

  // ---- stage weights in LDS, transposed, f16; fold 1/sqrt(H)=0.125 into Wq ----
  {
    const float* Ws[3] = {Wk, Wq, Wv};
    for (int idx = tid; idx < 3 * Hn * Cn; idx += 256) {
      int mat = idx / (Hn * Cn);
      int rem = idx - mat * (Hn * Cn);
      int h = rem / Cn;
      int c = rem - h * Cn;
      float w = Ws[mat][c * Hn + h];
      if (mat == 1) w *= 0.125f;
      wT[idx] = (_Float16)w;
    }
  }
  __syncthreads();

  // ================= Phase 1: K/Q/V = x @ W (f16 WMMA, f32 accum) =============
  for (int rti = 0; rti < 2; ++rti) {
    const int rt = wv + 8 * rti;                       // 16-row tile of T
    // A layout: lane(hi,m): element j -> K = 16*hi + j
    v16h afr[12];
    const float* xrow = xb + (size_t)(rt * 16 + n16) * Cn;
    #pragma unroll
    for (int kc = 0; kc < 12; ++kc) {
      const float4 f0 = *(const float4*)(xrow + kc * 32 + 16 * hi);
      const float4 f1 = *(const float4*)(xrow + kc * 32 + 16 * hi + 4);
      const float4 f2 = *(const float4*)(xrow + kc * 32 + 16 * hi + 8);
      const float4 f3 = *(const float4*)(xrow + kc * 32 + 16 * hi + 12);
      v16h a;
      a[0] = (_Float16)f0.x;  a[1] = (_Float16)f0.y;
      a[2] = (_Float16)f0.z;  a[3] = (_Float16)f0.w;
      a[4] = (_Float16)f1.x;  a[5] = (_Float16)f1.y;
      a[6] = (_Float16)f1.z;  a[7] = (_Float16)f1.w;
      a[8] = (_Float16)f2.x;  a[9] = (_Float16)f2.y;
      a[10] = (_Float16)f2.z; a[11] = (_Float16)f2.w;
      a[12] = (_Float16)f3.x; a[13] = (_Float16)f3.y;
      a[14] = (_Float16)f3.z; a[15] = (_Float16)f3.w;
      afr[kc] = a;
    }
    for (int mat = 0; mat < 3; ++mat) {
      for (int nc = 0; nc < 4; ++nc) {
        // B fragment: 16 contiguous f16 at wcol + kc*32 + 16*hi -> 2x ds_load_b128
        const _Float16* wcol = wT + (mat * Hn + nc * 16 + n16) * Cn + 16 * hi;
        v8f acc = {};
        v16h bcur = *(const v16h*)(wcol);
        #pragma unroll
        for (int kc = 0; kc < 12; ++kc) {
          v16h bnext;
          if (kc < 11) bnext = *(const v16h*)(wcol + (kc + 1) * 32);
          acc = wmma16(afr[kc], bcur, acc);
          bcur = bnext;
        }
        // Pin the pipeline: frag0+frag1 loads, then (wmma, next-frag loads) x10,
        // then the last two wmmas. 24 DS reads / 12 WMMAs total in region.
        SCHED_GROUP(0x100, 4, 0);
        #pragma unroll
        for (int i = 0; i < 10; ++i) {
          SCHED_GROUP(0x008, 1, 0);
          SCHED_GROUP(0x100, 2, 0);
        }
        SCHED_GROUP(0x008, 2, 0);
        // C layout: lane(hi,n): VGPR v holds row M = v + 8*hi, col n.
        #pragma unroll
        for (int v = 0; v < 8; ++v) {
          const int M = v + 8 * hi;
          const int row = rt * 16 + M;
          const int col = nc * 16 + n16;
          const _Float16 hvv = (_Float16)acc[v];
          if (mat == 0)      kr[row * Hn + col] = hvv;
          else if (mat == 1) qr[row * Hn + col] = hvv;
          else               vt[col * Tn + row] = hvv;   // store V transposed
        }
      }
    }
  }
  __syncthreads();

  // ================= Phase 2: causal flash attention ===========================
  for (int qi = 0; qi < 2; ++qi) {
    const int qt = (qi == 0) ? wv : (15 - wv);          // balanced tile pairing
    // Q A-fragments (16 x 64 = 2 frags of K=32)
    v16h qa[2];
    {
      const _Float16* qrow_p = qr + (qt * 16 + n16) * Hn;
      #pragma unroll
      for (int f = 0; f < 2; ++f) {
        const v8h lo  = *(const v8h*)(qrow_p + 32 * f + 16 * hi);
        const v8h hi8 = *(const v8h*)(qrow_p + 32 * f + 16 * hi + 8);
        qa[f] = cat8(lo, hi8);
      }
    }
    v8f o[4] = {v8f{}, v8f{}, v8f{}, v8f{}};
    float mrun[8], lrun[8];
    #pragma unroll
    for (int v = 0; v < 8; ++v) { mrun[v] = -1e30f; lrun[v] = 0.f; }

    const int kbmax = (qt * 16 + 15) >> 5;              // 32-key blocks
    for (int kb = 0; kb <= kbmax; ++kb) {
      // ---- all four K^T B-fragments (8x ds_load_b128), then 4 WMMAs ----
      v16h kb4[4];
      #pragma unroll
      for (int t = 0; t < 2; ++t) {
        const _Float16* krow_p = kr + (kb * 32 + t * 16 + n16) * Hn + 16 * hi;
        #pragma unroll
        for (int f = 0; f < 2; ++f)
          kb4[t * 2 + f] = *(const v16h*)(krow_p + 32 * f);
      }
      v8f s[2] = {v8f{}, v8f{}};
      s[0] = wmma16(qa[0], kb4[0], s[0]);
      s[0] = wmma16(qa[1], kb4[1], s[0]);
      s[1] = wmma16(qa[0], kb4[2], s[1]);
      s[1] = wmma16(qa[1], kb4[3], s[1]);
      SCHED_GROUP(0x100, 8, 1);
      SCHED_GROUP(0x008, 4, 1);

      // ---- causal mask + online softmax (row = one VGPR across 16 lanes) ----
      float p0[8], p1[8], alpha[8];
      #pragma unroll
      for (int v = 0; v < 8; ++v) {
        const int row = qt * 16 + v + 8 * hi;
        const int c0 = kb * 32 + n16;
        float a0 = (c0      <= row) ? s[0][v] : -1e30f;
        float a1 = (c0 + 16 <= row) ? s[1][v] : -1e30f;
        float mx = fmaxf(a0, a1);
        #pragma unroll
        for (int off = 1; off < 16; off <<= 1)
          mx = fmaxf(mx, __shfl_xor(mx, off, 32));
        const float mnew = fmaxf(mrun[v], mx);
        const float al = __expf(mrun[v] - mnew);
        const float e0 = __expf(a0 - mnew);
        const float e1 = __expf(a1 - mnew);
        float rs = e0 + e1;
        #pragma unroll
        for (int off = 1; off < 16; off <<= 1)
          rs += __shfl_xor(rs, off, 32);
        lrun[v] = lrun[v] * al + rs;
        mrun[v] = mnew;
        alpha[v] = al;
        p0[v] = e0; p1[v] = e1;
      }
      #pragma unroll
      for (int nc = 0; nc < 4; ++nc)
        #pragma unroll
        for (int v = 0; v < 8; ++v) o[nc][v] *= alpha[v];

      // ---- stage P (C-layout -> row-major LDS), then overlap V loads ----
      _Float16* pw = ps + wv * 512;
      #pragma unroll
      for (int v = 0; v < 8; ++v) {
        const int M = v + 8 * hi;
        pw[M * 32 + n16]      = (_Float16)p0[v];
        pw[M * 32 + n16 + 16] = (_Float16)p1[v];
      }
      // V B-fragments (independent LDS region; DS is in-order per wave)
      v16h vb[4];
      #pragma unroll
      for (int nc = 0; nc < 4; ++nc) {
        const _Float16* vcol = vt + (nc * 16 + n16) * Tn + kb * 32 + 16 * hi;
        vb[nc] = *(const v16h*)(vcol);
      }
      // Reload P as A-fragment
      v16h pa;
      {
        const _Float16* pr = pw + n16 * 32;
        const v8h lo  = *(const v8h*)(pr + 16 * hi);
        const v8h hi8 = *(const v8h*)(pr + 16 * hi + 8);
        pa = cat8(lo, hi8);
      }
      // ---- O += P V ----
      #pragma unroll
      for (int nc = 0; nc < 4; ++nc) o[nc] = wmma16(pa, vb[nc], o[nc]);
    }
    // ---- normalize and write out (fp32) ----
    float* ob = out + ((size_t)b * Tn + qt * 16) * Hn;
    #pragma unroll
    for (int v = 0; v < 8; ++v) {
      const int M = v + 8 * hi;
      const float inv = 1.f / lrun[v];
      ob[M * Hn +  0 + n16] = o[0][v] * inv;
      ob[M * Hn + 16 + n16] = o[1][v] * inv;
      ob[M * Hn + 32 + n16] = o[2][v] * inv;
      ob[M * Hn + 48 + n16] = o[3][v] * inv;
    }
  }
}

extern "C" void kernel_launch(void* const* d_in, const int* in_sizes, int n_in,
                              void* d_out, int out_size, void* d_ws, size_t ws_size,
                              hipStream_t stream) {
  (void)in_sizes; (void)n_in; (void)out_size; (void)d_ws; (void)ws_size;
  const float* x  = (const float*)d_in[0];
  const float* Wk = (const float*)d_in[1];
  const float* Wq = (const float*)d_in[2];
  const float* Wv = (const float*)d_in[3];
  float* out = (float*)d_out;
  (void)hipFuncSetAttribute((const void*)Head_attn_kernel,
                            hipFuncAttributeMaxDynamicSharedMemorySize,
                            (int)LDS_BYTES);
  Head_attn_kernel<<<1024, 256, LDS_BYTES, stream>>>(x, Wk, Wq, Wv, out);
}
